// SGDAT_43550968381554
// MI455X (gfx1250) — compile-verified, matrix operand order checked by reading, not run on hardware
//
#include <hip/hip_runtime.h>

typedef _Float16 h8  __attribute__((ext_vector_type(8)));
typedef _Float16 v16h __attribute__((ext_vector_type(16)));
typedef float    v8f  __attribute__((ext_vector_type(8)));

static constexpr int BB = 8, NN_PTS = 16384, DD = 64, KNN = 16, MM1 = 512, MM2 = 128, NC = 13;
static constexpr float LN_EPS = 1e-5f;

__device__ __forceinline__ float wave_sum(float v) {
  for (int o = 16; o > 0; o >>= 1) v += __shfl_xor(v, o, 32);
  return v;
}
__device__ __forceinline__ float silu_f(float x) { return x / (1.f + __expf(-x)); }

// Load a 16-half WMMA A/B fragment for one lane: two contiguous 16B loads.
// elements 0..7  <- K = kbase+koff .. +7
// elements 8..15 <- K = kbase+koff+16 .. +23   (koff = lane<16 ? 0 : 8)
__device__ __forceinline__ v16h load_frag(const _Float16* p) {
  h8 lo = *(const h8*)(p);
  h8 hi = *(const h8*)(p + 16);
  return __builtin_shufflevector(lo, hi, 0,1,2,3,4,5,6,7,8,9,10,11,12,13,14,15);
}

// ---------------------------------------------------------------- GEMM (WMMA)
// C[M,N] = A[M,Kp] (f16, row-major, lda) @ Wt[Npad,Kp]^T (f16, "B" stored as W^T)
// one wave per 16x16 tile; f32 accum; optional bias; optional f32/f16 stores.
__global__ void gemm_wmma_kernel(const _Float16* __restrict__ A, int lda,
                                 const _Float16* __restrict__ Wt, int ldw,
                                 const float* __restrict__ bias,
                                 float* __restrict__ Cf, int ldc,
                                 _Float16* __restrict__ Ch, int ldch,
                                 int M, int Kp, int Ntiles, int Nact) {
  const int lane = threadIdx.x & 31;
  const int wave = (blockIdx.x * blockDim.x + threadIdx.x) >> 5;
  const int Mtiles = M >> 4;
  if (wave >= Mtiles * Ntiles) return;
  const int tm = wave / Ntiles;
  const int tn = wave - tm * Ntiles;
  const int l15 = lane & 15;
  const int koff = (lane < 16) ? 0 : 8;
  const _Float16* Ap = A + (size_t)(tm * 16 + l15) * lda + koff;
  const _Float16* Bp = Wt + (size_t)(tn * 16 + l15) * ldw + koff;
  v8f acc = {0.f, 0.f, 0.f, 0.f, 0.f, 0.f, 0.f, 0.f};
  for (int k = 0; k < Kp; k += 32) {
    v16h a = load_frag(Ap + k);
    v16h b = load_frag(Bp + k);
    acc = __builtin_amdgcn_wmma_f32_16x16x32_f16(false, a, false, b, (short)0, acc,
                                                 false, false);
  }
  const int col = tn * 16 + l15;
  if (col < Nact) {
    const float bv = bias ? bias[col] : 0.f;
    const int rbase = tm * 16 + ((lane < 16) ? 0 : 8);
    #pragma unroll
    for (int r = 0; r < 8; ++r) {
      const int row = rbase + r;
      const float v = acc[r] + bv;
      if (Cf) Cf[(size_t)row * ldc + col] = v;
      if (Ch) Ch[(size_t)row * ldch + col] = (_Float16)v;
    }
  }
}

// ------------------------------------------------------- weight prep (W -> W^T f16, padded)
__global__ void prep_wt_kernel(const float* __restrict__ W, _Float16* __restrict__ Wt,
                               int K, int Nw, int Kp, int Npad) {
  int i = blockIdx.x * blockDim.x + threadIdx.x;
  if (i >= Npad * Kp) return;
  int n = i / Kp, k = i - n * Kp;
  float v = (k < K && n < Nw) ? W[(size_t)k * Nw + n] : 0.f;
  Wt[i] = (_Float16)v;
}

// ------------------------------------------------------- stem: 3 -> 64, SiLU, LN (wave/row)
__global__ void stem_kernel(const float* __restrict__ pts, const float* __restrict__ W,
                            const float* __restrict__ bv, const float* __restrict__ g,
                            const float* __restrict__ be, float* __restrict__ feats,
                            int rows) {
  const int lane = threadIdx.x & 31;
  const int row = (blockIdx.x * blockDim.x + threadIdx.x) >> 5;
  if (row >= rows) return;
  const float p0 = pts[(size_t)row * 3 + 0];
  const float p1 = pts[(size_t)row * 3 + 1];
  const float p2 = pts[(size_t)row * 3 + 2];
  const int f1 = lane + 32;
  float x0 = silu_f(p0 * W[lane] + p1 * W[64 + lane] + p2 * W[128 + lane] + bv[lane]);
  float x1 = silu_f(p0 * W[f1] + p1 * W[64 + f1] + p2 * W[128 + f1] + bv[f1]);
  float mu = wave_sum(x0 + x1) * (1.f / 64.f);
  float d0 = x0 - mu, d1 = x1 - mu;
  float var = wave_sum(d0 * d0 + d1 * d1) * (1.f / 64.f);
  float inv = rsqrtf(var + LN_EPS);
  feats[(size_t)row * 64 + lane] = d0 * inv * g[lane] + be[lane];
  feats[(size_t)row * 64 + f1]  = d1 * inv * g[f1] + be[f1];
}

// ------------------------------------------------------- SiLU + LayerNorm over 64 (wave/row)
__global__ void silu_ln_kernel(const float* __restrict__ in, const float* __restrict__ g,
                               const float* __restrict__ be, float* __restrict__ outf,
                               _Float16* __restrict__ outh, int rows) {
  const int lane = threadIdx.x & 31;
  const int row = (blockIdx.x * blockDim.x + threadIdx.x) >> 5;
  if (row >= rows) return;
  const size_t base = (size_t)row * 64;
  float x0 = silu_f(in[base + lane]);
  float x1 = silu_f(in[base + lane + 32]);
  float mu = wave_sum(x0 + x1) * (1.f / 64.f);
  float d0 = x0 - mu, d1 = x1 - mu;
  float var = wave_sum(d0 * d0 + d1 * d1) * (1.f / 64.f);
  float inv = rsqrtf(var + LN_EPS);
  float y0 = d0 * inv * g[lane] + be[lane];
  float y1 = d1 * inv * g[lane + 32] + be[lane + 32];
  if (outf) { outf[base + lane] = y0; outf[base + lane + 32] = y1; }
  if (outh) { outh[base + lane] = (_Float16)y0; outh[base + lane + 32] = (_Float16)y1; }
}

// ------------------------------------------------------- FPS: one block per batch, dist in LDS
__global__ void fps_kernel(const float* __restrict__ xyz, int Np, int Msel,
                           const int* __restrict__ baseIdx, int* __restrict__ outIdx,
                           float* __restrict__ outCent) {
  extern __shared__ char smem_raw[];
  const int b = blockIdx.x, tid = threadIdx.x, bd = blockDim.x;
  const float* P = xyz + (size_t)b * Np * 3;
  float* dist = (float*)smem_raw;          // Np floats
  float* red_d = dist + Np;                // bd floats
  int* red_i = (int*)(red_d + bd);         // bd ints
  for (int i = tid; i < Np; i += bd) dist[i] = 1e10f;
  __syncthreads();
  int far = 0;
  for (int m = 0; m < Msel; ++m) {
    if (tid == 0) {
      outIdx[b * Msel + m] = baseIdx ? baseIdx[b * Np + far] : far;
      outCent[((size_t)b * Msel + m) * 3 + 0] = P[far * 3 + 0];
      outCent[((size_t)b * Msel + m) * 3 + 1] = P[far * 3 + 1];
      outCent[((size_t)b * Msel + m) * 3 + 2] = P[far * 3 + 2];
    }
    const float cx = P[far * 3 + 0], cy = P[far * 3 + 1], cz = P[far * 3 + 2];
    float bestd = -1.f; int besti = 0x7fffffff;
    for (int i = tid; i < Np; i += bd) {
      float dx = P[i * 3] - cx, dy = P[i * 3 + 1] - cy, dz = P[i * 3 + 2] - cz;
      float nd = fminf(dist[i], dx * dx + dy * dy + dz * dz);
      dist[i] = nd;
      if (nd > bestd) { bestd = nd; besti = i; }
    }
    red_d[tid] = bestd; red_i[tid] = besti;
    __syncthreads();
    for (int s = bd >> 1; s > 0; s >>= 1) {
      if (tid < s) {
        float od = red_d[tid + s]; int oi = red_i[tid + s];
        if (od > red_d[tid] || (od == red_d[tid] && oi < red_i[tid])) {
          red_d[tid] = od; red_i[tid] = oi;
        }
      }
      __syncthreads();
    }
    far = red_i[0];
    __syncthreads();
  }
}

// ------------------------------------------------------- kNN: one block per center, top-16
__global__ void knn_kernel(const float* __restrict__ xyz, const float* __restrict__ cent,
                           int Np, int M, int* __restrict__ outIdx) {
  __shared__ float cd[128 * 16];
  __shared__ int   ci[128 * 16];
  __shared__ float rd[128];
  __shared__ int   rs[128];
  const int c = blockIdx.x;
  const int b = c / M;
  const int tid = threadIdx.x;
  const float* P = xyz + (size_t)b * Np * 3;
  const float cx = cent[(size_t)c * 3 + 0];
  const float cy = cent[(size_t)c * 3 + 1];
  const float cz = cent[(size_t)c * 3 + 2];
  float d16[16]; int i16[16];
  #pragma unroll
  for (int j = 0; j < 16; ++j) { d16[j] = 3e38f; i16[j] = 0x7fffffff; }
  for (int i = tid; i < Np; i += 128) {
    float dx = P[i * 3] - cx, dy = P[i * 3 + 1] - cy, dz = P[i * 3 + 2] - cz;
    float d = dx * dx + dy * dy + dz * dz;
    if (d < d16[15]) {
      int j = 15;
      while (j > 0 && d16[j - 1] > d) { d16[j] = d16[j - 1]; i16[j] = i16[j - 1]; --j; }
      d16[j] = d; i16[j] = i;
    }
  }
  for (int j = 0; j < 16; ++j) { cd[tid * 16 + j] = d16[j]; ci[tid * 16 + j] = i16[j]; }
  __syncthreads();
  for (int sel = 0; sel < 16; ++sel) {
    float bd = 3e38f; int bs = tid * 16;
    for (int j = 0; j < 16; ++j) {
      float v = cd[tid * 16 + j];
      if (v < bd) { bd = v; bs = tid * 16 + j; }
    }
    rd[tid] = bd; rs[tid] = bs;
    __syncthreads();
    for (int s = 64; s > 0; s >>= 1) {
      if (tid < s) {
        if (rd[tid + s] < rd[tid] ||
            (rd[tid + s] == rd[tid] && ci[rs[tid + s]] < ci[rs[tid]])) {
          rd[tid] = rd[tid + s]; rs[tid] = rs[tid + s];
        }
      }
      __syncthreads();
    }
    if (tid == 0) { outIdx[(size_t)c * 16 + sel] = ci[rs[0]]; cd[rs[0]] = 3e38f; }
    __syncthreads();
  }
}

// ------------------------------------------------------- nearest center per point
__global__ void nn_kernel(const float* __restrict__ xyz, const float* __restrict__ cent,
                          int Np, int M, int* __restrict__ out, int total) {
  int i = blockIdx.x * blockDim.x + threadIdx.x;
  if (i >= total) return;
  int b = i / Np;
  float px = xyz[(size_t)i * 3 + 0], py = xyz[(size_t)i * 3 + 1], pz = xyz[(size_t)i * 3 + 2];
  float bd = 3e38f; int bi = 0;
  for (int m = 0; m < M; ++m) {
    const float* cc = cent + ((size_t)b * M + m) * 3;
    float dx = px - cc[0], dy = py - cc[1], dz = pz - cc[2];
    float d = dx * dx + dy * dy + dz * dz;
    if (d < bd) { bd = d; bi = m; }
  }
  out[i] = bi;
}

// ------------------------------------------------------- build fuse input rows [.,160] f16
__global__ void build_fuse_kernel(const float* __restrict__ xyz, const float* __restrict__ feats,
                                  const float* __restrict__ cent, const int* __restrict__ knn,
                                  int M, _Float16* __restrict__ out, size_t total) {
  for (size_t idx = blockIdx.x * (size_t)blockDim.x + threadIdx.x; idx < total;
       idx += (size_t)gridDim.x * blockDim.x) {
    int f = (int)(idx % 160);
    size_t r = idx / 160;
    int j = (int)(r & 15);
    int cm = (int)(r >> 4);
    int b = cm / M;
    int nb = knn[(size_t)cm * 16 + j];
    int n0 = knn[(size_t)cm * 16];
    float v = 0.f;
    if (f < 64) {
      v = feats[((size_t)b * NN_PTS + n0) * 64 + f];
    } else if (f < 128) {
      int ff = f - 64;
      v = feats[((size_t)b * NN_PTS + nb) * 64 + ff] -
          feats[((size_t)b * NN_PTS + n0) * 64 + ff];
    } else if (f < 131) {
      int cc = f - 128;
      v = xyz[((size_t)b * NN_PTS + nb) * 3 + cc] - cent[(size_t)cm * 3 + cc];
    }
    out[idx] = (_Float16)v;
  }
}

// ------------------------------------------------------- mean over k=16 neighbor rows
__global__ void mean_k_kernel(const float* __restrict__ in, float* __restrict__ outf,
                              _Float16* __restrict__ outh, int rowsOut) {
  int i = blockIdx.x * blockDim.x + threadIdx.x;
  if (i >= rowsOut * 64) return;
  int r = i >> 6, f = i & 63;
  float s = 0.f;
  for (int j = 0; j < 16; ++j) s += in[(((size_t)r * 16) + j) * 64 + f];
  s *= (1.f / 16.f);
  if (outf) outf[i] = s;
  if (outh) outh[i] = (_Float16)s;
}

// ------------------------------------------------------- concat [feats, up2] -> f16 [.,128]
__global__ void build_bp_kernel(const float* __restrict__ feats, const float* __restrict__ out2,
                                const int* __restrict__ nn2, _Float16* __restrict__ out,
                                size_t total) {
  for (size_t idx = blockIdx.x * (size_t)blockDim.x + threadIdx.x; idx < total;
       idx += (size_t)gridDim.x * blockDim.x) {
    int f = (int)(idx & 127);
    size_t r = idx >> 7;
    int b = (int)(r / NN_PTS);
    float v = (f < 64) ? feats[r * 64 + f]
                       : out2[(((size_t)b * MM2) + nn2[r]) * 64 + (f - 64)];
    out[idx] = (_Float16)v;
  }
}

__global__ void idx2float_kernel(const int* __restrict__ idx, float* __restrict__ out, int n) {
  int i = blockIdx.x * blockDim.x + threadIdx.x;
  if (i < n) out[i] = (float)idx[i];
}

// =============================================================================
extern "C" void kernel_launch(void* const* d_in, const int* in_sizes, int n_in,
                              void* d_out, int out_size, void* d_ws, size_t ws_size,
                              hipStream_t stream) {
  (void)in_sizes; (void)n_in; (void)out_size; (void)ws_size;
  const float* points  = (const float*)d_in[0];
  const float* stem_w  = (const float*)d_in[1];
  const float* stem_b  = (const float*)d_in[2];
  const float* stem_g  = (const float*)d_in[3];
  const float* stem_be = (const float*)d_in[4];
  const float* f1_w1 = (const float*)d_in[5],  *f1_b1 = (const float*)d_in[6];
  const float* f1_g1 = (const float*)d_in[7],  *f1_be1 = (const float*)d_in[8];
  const float* f1_w2 = (const float*)d_in[9],  *f1_b2 = (const float*)d_in[10];
  const float* f1_g2 = (const float*)d_in[11], *f1_be2 = (const float*)d_in[12];
  const float* f1_pw = (const float*)d_in[13], *f1_pb = (const float*)d_in[14];
  const float* f2_w1 = (const float*)d_in[15], *f2_b1 = (const float*)d_in[16];
  const float* f2_g1 = (const float*)d_in[17], *f2_be1 = (const float*)d_in[18];
  const float* f2_w2 = (const float*)d_in[19], *f2_b2 = (const float*)d_in[20];
  const float* f2_g2 = (const float*)d_in[21], *f2_be2 = (const float*)d_in[22];
  const float* f2_pw = (const float*)d_in[23], *f2_pb = (const float*)d_in[24];
  const float* sem1_w = (const float*)d_in[25], *sem1_b = (const float*)d_in[26];
  const float* bp_w = (const float*)d_in[27], *bp_b = (const float*)d_in[28];
  const float* bp_g = (const float*)d_in[29], *bp_be = (const float*)d_in[30];
  const float* mh_w1 = (const float*)d_in[31], *mh_b1 = (const float*)d_in[32];
  const float* mh_g = (const float*)d_in[33], *mh_be = (const float*)d_in[34];
  const float* mh_w2 = (const float*)d_in[35], *mh_b2 = (const float*)d_in[36];

  char* ws = (char*)d_ws;
  size_t off = 0;
  auto alloc = [&](size_t bytes) -> void* {
    void* p = ws + off;
    off = (off + bytes + 255) & ~(size_t)255;
    return p;
  };
  float*     feats   = (float*)alloc((size_t)BB * NN_PTS * DD * 4);
  _Float16*  fuse_in = (_Float16*)alloc((size_t)BB * MM1 * KNN * 160 * 2);
  float*     raw     = (float*)alloc((size_t)BB * NN_PTS * DD * 4);
  float*     act     = (float*)alloc((size_t)BB * MM1 * KNN * DD * 4);
  _Float16*  hh      = (_Float16*)alloc((size_t)BB * NN_PTS * DD * 2);
  _Float16*  mean_h  = (_Float16*)alloc((size_t)BB * MM1 * DD * 2);
  _Float16*  out1_h  = (_Float16*)alloc((size_t)BB * MM1 * DD * 2);
  float*     out2_f  = (float*)alloc((size_t)BB * MM2 * DD * 4);
  _Float16*  bp_in   = (_Float16*)alloc((size_t)BB * NN_PTS * 128 * 2);
  int*       idx1    = (int*)alloc((size_t)BB * MM1 * 4);
  int*       idx2    = (int*)alloc((size_t)BB * MM2 * 4);
  float*     cent1   = (float*)alloc((size_t)BB * MM1 * 3 * 4);
  float*     cent2   = (float*)alloc((size_t)BB * MM2 * 3 * 4);
  int*       knn1    = (int*)alloc((size_t)BB * MM1 * KNN * 4);
  int*       knn2    = (int*)alloc((size_t)BB * MM2 * KNN * 4);
  int*       nn2v    = (int*)alloc((size_t)BB * NN_PTS * 4);
  _Float16*  w1t_1 = (_Float16*)alloc(64 * 160 * 2);
  _Float16*  w2t_1 = (_Float16*)alloc(64 * 64 * 2);
  _Float16*  pwt_1 = (_Float16*)alloc(64 * 64 * 2);
  _Float16*  w1t_2 = (_Float16*)alloc(64 * 160 * 2);
  _Float16*  w2t_2 = (_Float16*)alloc(64 * 64 * 2);
  _Float16*  pwt_2 = (_Float16*)alloc(64 * 64 * 2);
  _Float16*  semwt = (_Float16*)alloc(16 * 64 * 2);
  _Float16*  bpwt  = (_Float16*)alloc(64 * 128 * 2);
  _Float16*  mh1t  = (_Float16*)alloc(64 * 64 * 2);
  _Float16*  mh2t  = (_Float16*)alloc(16 * 64 * 2);

  float* outF = (float*)d_out;
  float* out_logits = outF;                                   // (B,N,13)
  float* out_sem    = outF + (size_t)BB * NN_PTS * NC;        // (B,512,13)
  float* out_idx1   = out_sem + (size_t)BB * MM1 * NC;        // (B,512)
  float* out_idx2   = out_idx1 + (size_t)BB * MM1;            // (B,128)

  auto prep = [&](const float* W, _Float16* Wt, int K, int Nw, int Kp, int Npad) {
    int total = Npad * Kp;
    prep_wt_kernel<<<(total + 255) / 256, 256, 0, stream>>>(W, Wt, K, Nw, Kp, Npad);
  };
  auto gemm = [&](const _Float16* A, int lda, const _Float16* Wt, int ldw,
                  const float* bias, float* Cf, int ldc, _Float16* Ch, int ldch,
                  int M, int Kp, int Ntiles, int Nact) {
    int waves = (M >> 4) * Ntiles;
    int blocks = (waves + 7) / 8;
    gemm_wmma_kernel<<<blocks, 256, 0, stream>>>(A, lda, Wt, ldw, bias, Cf, ldc,
                                                 Ch, ldch, M, Kp, Ntiles, Nact);
  };

  // weight transposes -> f16
  prep(f1_w1, w1t_1, 131, 64, 160, 64);
  prep(f1_w2, w2t_1, 64, 64, 64, 64);
  prep(f1_pw, pwt_1, 64, 64, 64, 64);
  prep(f2_w1, w1t_2, 131, 64, 160, 64);
  prep(f2_w2, w2t_2, 64, 64, 64, 64);
  prep(f2_pw, pwt_2, 64, 64, 64, 64);
  prep(sem1_w, semwt, 64, 13, 64, 16);
  prep(bp_w, bpwt, 128, 64, 128, 64);
  prep(mh_w1, mh1t, 64, 64, 64, 64);
  prep(mh_w2, mh2t, 64, 13, 64, 16);

  // stem
  stem_kernel<<<(BB * NN_PTS + 7) / 8, 256, 0, stream>>>(points, stem_w, stem_b,
                                                         stem_g, stem_be, feats,
                                                         BB * NN_PTS);
  // FPS stage 1 (dist array in LDS: 64KB + reduction scratch)
  fps_kernel<<<BB, 1024, NN_PTS * 4 + 1024 * 8, stream>>>(points, NN_PTS, MM1,
                                                          nullptr, idx1, cent1);
  knn_kernel<<<BB * MM1, 128, 0, stream>>>(points, cent1, NN_PTS, MM1, knn1);
  {
    size_t total = (size_t)BB * MM1 * KNN * 160;
    build_fuse_kernel<<<(int)((total + 255) / 256), 256, 0, stream>>>(
        points, feats, cent1, knn1, MM1, fuse_in, total);
  }
  // fuse-1 MLP (WMMA): 160->64 SiLU LN, 64->64 SiLU LN, mean(k), proj, sem head
  gemm(fuse_in, 160, w1t_1, 160, f1_b1, raw, 64, nullptr, 0, BB * MM1 * KNN, 160, 4, 64);
  silu_ln_kernel<<<(BB * MM1 * KNN + 7) / 8, 256, 0, stream>>>(raw, f1_g1, f1_be1,
                                                               nullptr, hh, BB * MM1 * KNN);
  gemm(hh, 64, w2t_1, 64, f1_b2, raw, 64, nullptr, 0, BB * MM1 * KNN, 64, 4, 64);
  silu_ln_kernel<<<(BB * MM1 * KNN + 7) / 8, 256, 0, stream>>>(raw, f1_g2, f1_be2,
                                                               act, nullptr, BB * MM1 * KNN);
  mean_k_kernel<<<(BB * MM1 * 64 + 255) / 256, 256, 0, stream>>>(act, nullptr, mean_h,
                                                                 BB * MM1);
  gemm(mean_h, 64, pwt_1, 64, f1_pb, nullptr, 0, out1_h, 64, BB * MM1, 64, 4, 64);
  gemm(out1_h, 64, semwt, 64, sem1_b, out_sem, 13, nullptr, 0, BB * MM1, 64, 1, 13);

  // FPS stage 2 on centers1, mapped through idx1
  fps_kernel<<<BB, 512, MM1 * 4 + 512 * 8, stream>>>(cent1, MM1, MM2, idx1, idx2, cent2);
  knn_kernel<<<BB * MM2, 128, 0, stream>>>(points, cent2, NN_PTS, MM2, knn2);
  {
    size_t total = (size_t)BB * MM2 * KNN * 160;
    build_fuse_kernel<<<(int)((total + 255) / 256), 256, 0, stream>>>(
        points, feats, cent2, knn2, MM2, fuse_in, total);
  }
  gemm(fuse_in, 160, w1t_2, 160, f2_b1, raw, 64, nullptr, 0, BB * MM2 * KNN, 160, 4, 64);
  silu_ln_kernel<<<(BB * MM2 * KNN + 7) / 8, 256, 0, stream>>>(raw, f2_g1, f2_be1,
                                                               nullptr, hh, BB * MM2 * KNN);
  gemm(hh, 64, w2t_2, 64, f2_b2, raw, 64, nullptr, 0, BB * MM2 * KNN, 64, 4, 64);
  silu_ln_kernel<<<(BB * MM2 * KNN + 7) / 8, 256, 0, stream>>>(raw, f2_g2, f2_be2,
                                                               act, nullptr, BB * MM2 * KNN);
  mean_k_kernel<<<(BB * MM2 * 64 + 255) / 256, 256, 0, stream>>>(act, nullptr, mean_h,
                                                                 BB * MM2);
  gemm(mean_h, 64, pwt_2, 64, f2_pb, out2_f, 64, nullptr, 0, BB * MM2, 64, 4, 64);

  // backbone: upsample + concat + MLP head
  nn_kernel<<<(BB * NN_PTS + 255) / 256, 256, 0, stream>>>(points, cent2, NN_PTS, MM2,
                                                           nn2v, BB * NN_PTS);
  {
    size_t total = (size_t)BB * NN_PTS * 128;
    build_bp_kernel<<<(int)((total + 255) / 256), 256, 0, stream>>>(feats, out2_f, nn2v,
                                                                    bp_in, total);
  }
  gemm(bp_in, 128, bpwt, 128, bp_b, raw, 64, nullptr, 0, BB * NN_PTS, 128, 4, 64);
  silu_ln_kernel<<<(BB * NN_PTS + 7) / 8, 256, 0, stream>>>(raw, bp_g, bp_be, nullptr,
                                                            hh, BB * NN_PTS);
  gemm(hh, 64, mh1t, 64, mh_b1, raw, 64, nullptr, 0, BB * NN_PTS, 64, 4, 64);
  silu_ln_kernel<<<(BB * NN_PTS + 7) / 8, 256, 0, stream>>>(raw, mh_g, mh_be, nullptr,
                                                            hh, BB * NN_PTS);
  gemm(hh, 64, mh2t, 64, mh_b2, out_logits, 13, nullptr, 0, BB * NN_PTS, 64, 1, 13);

  // index outputs as floats
  idx2float_kernel<<<(BB * MM1 + 255) / 256, 256, 0, stream>>>(idx1, out_idx1, BB * MM1);
  idx2float_kernel<<<(BB * MM2 + 255) / 256, 256, 0, stream>>>(idx2, out_idx2, BB * MM2);
}
